// Conv2D_79594333929943
// MI455X (gfx1250) — matrix operand order repaired and checked by reading, not compiled, not run on hardware
//
#include <hip/hip_runtime.h>

typedef __attribute__((ext_vector_type(16))) _Float16 v16h;
typedef __attribute__((ext_vector_type(8)))  _Float16 v8h;
typedef __attribute__((ext_vector_type(8)))  float    v8f;
typedef __attribute__((ext_vector_type(4)))  float    v4f;

#define C_IN  64
#define K_OUT 128
#define HW    128

// ---------------------------------------------------------------------------
// Prep kernel: weights OIHW fp32 -> W2[rs][kout][c] f16 (c contiguous).
// 3*3*128*64 = 73728 elements = 288 blocks * 256 threads.
// ---------------------------------------------------------------------------
__global__ void conv_prep_weights(const float* __restrict__ w,
                                  _Float16* __restrict__ w2) {
    int idx = blockIdx.x * 256 + threadIdx.x;       // < 73728
    int c    = idx & 63;
    int kout = (idx >> 6) & 127;
    int rs   = idx >> 13;                           // 0..8
    int r = rs / 3, s = rs % 3;
    float val = w[((size_t)(kout * C_IN + c) * 3 + r) * 3 + s];
    w2[((size_t)rs * K_OUT + kout) * C_IN + c] = (_Float16)val;
}

// ---------------------------------------------------------------------------
// Main kernel: implicit-GEMM conv via v_wmma_f32_16x16x32_f16.
// Block = 256 threads (8 waves). Each block: 32 spatial (one row segment) x
// all 128 out-channels. Wave w -> kout slab [16w,16w+16), two 16x16 m-tiles.
// ---------------------------------------------------------------------------
__global__ void __launch_bounds__(256)
conv_main(const float* __restrict__ x, const _Float16* __restrict__ w2,
          const float* __restrict__ bias, float* __restrict__ out) {
    // Input patch: [py 0..2][px 0..33][c 0..63], f16, c-contiguous (128B rows)
    __shared__ alignas(64) _Float16 Xs[3][34][C_IN];

    const int bx = blockIdx.x;
    const int w0 = (bx & 3) * 32;          // 4 tiles per image row
    const int h  = (bx >> 2) & 127;
    const int n  = bx >> 9;                // 4*128 = 512 blocks per image

    const int tid = threadIdx.x;

    // ---- Stage input patch (with zero halo) into LDS, fp32 -> f16 ----------
    const float* xin = x + (size_t)n * C_IN * HW * HW;
    for (int idx = tid; idx < 3 * 34 * C_IN; idx += 256) {
        int px = idx % 34;                 // consecutive lanes -> consecutive x
        int t  = idx / 34;
        int py = t % 3;
        int c  = t / 3;
        int gy = h - 1 + py;
        int gx = w0 - 1 + px;
        float v = 0.0f;
        if ((unsigned)gy < 128u && (unsigned)gx < 128u)
            v = xin[((size_t)c * HW + gy) * HW + gx];
        Xs[py][px][c] = (_Float16)v;
    }
    __syncthreads();

    const int lane = tid & 31;
    const int wave = tid >> 5;
    const int nb   = lane & 15;            // column within the 16-wide tile
    const int hiL  = lane >> 4;            // 0 for lanes 0-15, 1 for 16-31
    const int kout = wave * 16 + nb;

    v8f acc0 = {};                         // spatial rows 0..15  of the tile
    v8f acc1 = {};                         // spatial rows 16..31 of the tile

    // ---- K loop: 9 filter taps x 2 k-steps of 32 channels ------------------
    for (int rs = 0; rs < 9; ++rs) {
        const int r = rs / 3, s = rs % 3;
        const _Float16* wb = w2 + ((size_t)rs * K_OUT + kout) * C_IN;
        #pragma unroll
        for (int kk = 0; kk < 2; ++kk) {
            const int kbase = kk * 32;

            // B fragment (32x16 f16): lane = column; halves 0..7 hold
            // K = kbase + (lane>=16?8:0) + 0..7, halves 8..15 the +16 chunk.
            v16h bfrag;
            ((v8h*)&bfrag)[0] = *(const v8h*)(wb + kbase + hiL * 8);
            ((v8h*)&bfrag)[1] = *(const v8h*)(wb + kbase + 16 + hiL * 8);

            // A fragment (16x32 f16): lane%16 = M row; halves 0..7 hold
            // K = kbase + (lane>=16?8:0) + 0..7, halves 8..15 the +16 chunk.
            const int c0 = kbase + hiL * 8;
            const _Float16* a0p = &Xs[r][nb + s][c0];
            const _Float16* a1p = &Xs[r][nb + 16 + s][c0];
            v16h a0, a1;
            ((v8h*)&a0)[0] = *(const v8h*)(a0p);
            ((v8h*)&a0)[1] = *(const v8h*)(a0p + 16);
            ((v8h*)&a1)[0] = *(const v8h*)(a1p);
            ((v8h*)&a1)[1] = *(const v8h*)(a1p + 16);

            acc0 = __builtin_amdgcn_wmma_f32_16x16x32_f16(
                       false, a0, false, bfrag, (short)0, acc0, false, false);
            acc1 = __builtin_amdgcn_wmma_f32_16x16x32_f16(
                       false, a1, false, bfrag, (short)0, acc1, false, false);
        }
    }

    // ---- Bias + store ------------------------------------------------------
    const float bv = bias[kout];
    acc0 += bv;
    acc1 += bv;

    // C/D layout: element e -> M = e + (lane>=16 ? 8 : 0); N = lane%16.
    // Per lane: 8 consecutive w positions per accumulator -> 2x b128 stores.
    const int mrow = hiL * 8;
    float* op = out + (((size_t)(n * K_OUT + kout)) * HW + h) * HW + w0 + mrow;
    *(v4f*)(op +  0) = ((v4f*)&acc0)[0];
    *(v4f*)(op +  4) = ((v4f*)&acc0)[1];
    *(v4f*)(op + 16) = ((v4f*)&acc1)[0];
    *(v4f*)(op + 20) = ((v4f*)&acc1)[1];
}

// ---------------------------------------------------------------------------
extern "C" void kernel_launch(void* const* d_in, const int* in_sizes, int n_in,
                              void* d_out, int out_size, void* d_ws, size_t ws_size,
                              hipStream_t stream) {
    const float* x    = (const float*)d_in[0];   // (16,64,128,128) fp32
    const float* w    = (const float*)d_in[1];   // (128,64,3,3)    fp32
    const float* bias = (const float*)d_in[2];   // (128,)          fp32
    float* out = (float*)d_out;                  // (16,128,128,128) fp32

    _Float16* w2 = (_Float16*)d_ws;              // 73728 halves = 144 KB

    conv_prep_weights<<<dim3(288), dim3(256), 0, stream>>>(w, w2);
    conv_main<<<dim3(8192), dim3(256), 0, stream>>>(x, w2, bias, out);
}
